// LSTM_16157666967933
// MI455X (gfx1250) — matrix operand order repaired and checked by reading, not compiled, not run on hardware
//
#include <hip/hip_runtime.h>

// ---------------------------------------------------------------------------
// Problem constants (match the reference).
// ---------------------------------------------------------------------------
#define T_STEPS 2048
#define BATCH   128
#define IN_DIM  64
#define H_DIM   128
#define G_DIM   512          // 4*H, gate order i,f,g,o
#define OUT_DIM 32

// ---------------------------------------------------------------------------
// Vector types for the CDNA5 WMMA builtins (wave32).
// ---------------------------------------------------------------------------
typedef __attribute__((ext_vector_type(4)))  __bf16 v4bf;
typedef __attribute__((ext_vector_type(8)))  __bf16 v8bf;
typedef __attribute__((ext_vector_type(16))) __bf16 v16bf;
typedef __attribute__((ext_vector_type(4)))  float  v4f;
typedef __attribute__((ext_vector_type(8)))  float  v8f;

__device__ __forceinline__ __bf16 to_bf16(float f) {
  // round-to-nearest-even f32 -> bf16
  unsigned int u = __builtin_bit_cast(unsigned int, f);
  u += 0x7FFFu + ((u >> 16) & 1u);
  unsigned short s = (unsigned short)(u >> 16);
  return __builtin_bit_cast(__bf16, s);
}

// gfx1250 has a TRANS-unit v_tanh_f32; TRANS ops co-execute with WMMA.
__device__ __forceinline__ float fast_tanh(float x) {
#if __has_builtin(__builtin_amdgcn_tanhf)
  return __builtin_amdgcn_tanhf(x);
#elif __has_builtin(__builtin_amdgcn_tanh_f32)
  return __builtin_amdgcn_tanh_f32(x);
#else
  float e = __expf(2.0f * x);
  return 1.0f - 2.0f / (e + 1.0f);
#endif
}
// Exact identity: sigma(x) = 0.5*tanh(x/2) + 0.5  (one TRANS op per gate)
__device__ __forceinline__ float fast_sigmoid(float x) {
  return __builtin_fmaf(0.5f, fast_tanh(0.5f * x), 0.5f);
}

__device__ __forceinline__ v16bf cat16(v8bf lo, v8bf hi) {
  return __builtin_shufflevector(lo, hi, 0, 1, 2, 3, 4, 5, 6, 7,
                                         8, 9, 10, 11, 12, 13, 14, 15);
}

// A-matrix fragment (16x32 bf16, M x K), A stored row-major [16][ktot].
// lanes 0-15 -> row M=lane, K = {kb..kb+7, kb+16..kb+23}
// lanes 16-31 -> row M=lane-16, K = {kb+8..kb+15, kb+24..kb+31}
__device__ __forceinline__ v16bf load_a_frag(const __bf16* A, int ktot,
                                             int kbase, int lane) {
  int row = lane & 15;
  int klo = kbase + ((lane >> 4) << 3);          // +0 or +8
  const __bf16* p = A + row * ktot + klo;
  v8bf lo = *(const v8bf*)p;
  v8bf hi = *(const v8bf*)(p + 16);
  return cat16(lo, hi);
}

// B-matrix fragment (32x16 bf16, K x N). W stored as [N][ktot] row-major
// (the torch weight layout), so B[k][n] = W[n][k].
// lanes 0-15 -> col N=lane, K = {kb..kb+7, kb+16..kb+23} ; lanes 16-31 -> K = {kb+8..kb+15, kb+24..kb+31}
__device__ __forceinline__ v16bf load_b_frag(const __bf16* W, int ktot,
                                             int nbase, int kbase, int lane) {
  int col = nbase + (lane & 15);
  int k0  = kbase + ((lane >> 4) << 3);          // +0 or +8
  const __bf16* p = W + col * ktot + k0;
  v8bf lo = *(const v8bf*)p;
  v8bf hi = *(const v8bf*)(p + 16);
  return cat16(lo, hi);
}

// ---------------------------------------------------------------------------
// CDNA5 async global->LDS copies (ISA 15.18.3 opcodes 97/98, ASYNCcnt).
// vdst VGPR carries the per-lane LDS byte address.
// ---------------------------------------------------------------------------
__device__ __forceinline__ void async_copy_b128(unsigned lds_addr, const void* g) {
  asm volatile("global_load_async_to_lds_b128 %0, %1, off"
               :: "v"(lds_addr), "v"((unsigned long long)(uintptr_t)g)
               : "memory");
}
__device__ __forceinline__ void async_copy_b64(unsigned lds_addr, const void* g) {
  asm volatile("global_load_async_to_lds_b64 %0, %1, off"
               :: "v"(lds_addr), "v"((unsigned long long)(uintptr_t)g)
               : "memory");
}
__device__ __forceinline__ void wait_async0() {
#if __has_builtin(__builtin_amdgcn_s_wait_asynccnt)
  __builtin_amdgcn_s_wait_asynccnt(0);
#else
  asm volatile("s_wait_asynccnt 0x0" ::: "memory");
#endif
}

// ---------------------------------------------------------------------------
// x (f32) -> bf16 pre-convert, fully parallel, so both scan layers can stage
// their inputs as raw byte copies through the async-to-LDS path.
// ---------------------------------------------------------------------------
__global__ __launch_bounds__(256) void cvt_kernel(const float* __restrict__ in,
                                                  __bf16* __restrict__ out,
                                                  int n4) {
  int i = blockIdx.x * 256 + threadIdx.x;
  if (i >= n4) return;
  v4f f = *((const v4f*)in + i);
  v4bf o;
#pragma unroll
  for (int j = 0; j < 4; ++j) o[j] = to_bf16(f[j]);
  *((v4bf*)out + i) = o;
}

// ---------------------------------------------------------------------------
// Persistent LSTM layer scan. One workgroup owns 16 batch rows for all T
// steps (batch rows independent -> zero inter-WG sync).
//   grid = BATCH/16 = 8 blocks, block = 256 threads = 8 waves (wave32).
// gates[16,512] = [h | x_t] x [w_hh | w_ih]^T.  Wave w computes the four
// 16x16 gate tiles at N = {0,128,256,384} + w*16, so i/f/g/o for one
// (batch,h-col) land in the same lane; cell state c stays in registers.
// w_hh B-fragments (4 gates x 4 K-chunks = 128 VGPRs) are register-resident
// for the whole scan; x_t is double-buffered via async-to-LDS one step ahead.
// ---------------------------------------------------------------------------
template <int KX>
__global__ __launch_bounds__(256) void lstm_scan_kernel(
    const __bf16* __restrict__ x,         // [T,B,KX] bf16
    const float*  __restrict__ w_ih,      // [512, KX]
    const float*  __restrict__ w_hh,      // [512, H]
    __bf16*       __restrict__ y_out,     // [T,B,H] bf16
    float*        __restrict__ hc_out,    // d_out + T*B*OUT: h_n then c_n
    int layer)
{
  constexpr int KTOT    = H_DIM + KX;       // 192 / 256
  constexpr int KCX     = KX / 32;          // x-part K-chunks: 2 / 4
  constexpr int HT_OFF  = G_DIM * KTOT * 2; // bytes
  constexpr int XB_OFF  = HT_OFF + 16 * H_DIM * 2;
  constexpr int XB_BYTES = 16 * KX * 2;     // one x_t slab (2 KB / 4 KB)

  extern __shared__ char smem[];
  __bf16* Wl = (__bf16*)smem;               // [512][KTOT] fused [w_hh|w_ih]
  __bf16* Ht = (__bf16*)(smem + HT_OFF);    // [16][128] h tile
  __bf16* XB = (__bf16*)(smem + XB_OFF);    // [2][16*KX] x double buffer

  const int tid  = threadIdx.x;
  const int lane = tid & 31;
  const int wv   = tid >> 5;                // 0..7
  const int b0   = blockIdx.x * 16;

  // ---- preload fused weights (bf16) and zero h0 ---------------------------
  for (int idx = tid; idx < G_DIM * KTOT; idx += 256) {
    int n = idx / KTOT;
    int k = idx - n * KTOT;
    float w = (k < H_DIM) ? w_hh[n * H_DIM + k] : w_ih[n * KX + (k - H_DIM)];
    Wl[idx] = to_bf16(w);
  }
  for (int idx = tid; idx < 16 * H_DIM; idx += 256)
    Ht[idx] = to_bf16(0.0f);

  // ---- async-stage x_0 into XB[0] (retired at top of step 0) --------------
  const unsigned xb_lds = __builtin_amdgcn_groupstaticsize() + (unsigned)XB_OFF;
  auto stage_async = [&](int t, int buf) {
    const char* slab = (const char*)(x + ((size_t)t * BATCH + b0) * KX);
    unsigned dst = xb_lds + (unsigned)buf * XB_BYTES;
    if constexpr (KX == 64)  async_copy_b64 (dst + tid * 8,  slab + tid * 8);
    else                     async_copy_b128(dst + tid * 16, slab + tid * 16);
  };
  stage_async(0, 0);

  __syncthreads();   // weights + h0 visible

  // ---- register-resident w_hh B-fragments (4 gates x 4 chunks) ------------
  v16bf bhh[4][4];
#pragma unroll
  for (int g4 = 0; g4 < 4; ++g4)
#pragma unroll
    for (int kc = 0; kc < 4; ++kc)
      bhh[g4][kc] = load_b_frag(Wl, KTOT, g4 * H_DIM + wv * 16, kc * 32, lane);

  // Register state: (VGPR r, lane) -> (m = r + mrow, hcol).
  const int mrow = (lane >> 4) << 3;        // 0 or 8
  const int hcol = wv * 16 + (lane & 15);
  float c_st[8], h_f32[8];
#pragma unroll
  for (int r = 0; r < 8; ++r) { c_st[r] = 0.0f; h_f32[r] = 0.0f; }

  for (int t = 0; t < T_STEPS; ++t) {
    const int cur = t & 1;
    wait_async0();       // my async writes of XB[cur] are done
    __syncthreads();     // everyone's XB[cur] + h(t) visible

    if (t + 1 < T_STEPS) stage_async(t + 1, cur ^ 1);

    // ---- recurrent GEMM (one A-frag live at a time) -----------------------
    v8f acc[4] = {v8f{}, v8f{}, v8f{}, v8f{}};
#pragma unroll
    for (int kc = 0; kc < 4; ++kc) {                 // h part: B from regs
      v16bf a = load_a_frag(Ht, H_DIM, kc * 32, lane);
#pragma unroll
      for (int g4 = 0; g4 < 4; ++g4)
        acc[g4] = __builtin_amdgcn_wmma_f32_16x16x32_bf16(
                      false, a, false, bhh[g4][kc], (short)0, acc[g4],
                      false, false);
    }
    const __bf16* XBc = XB + cur * (16 * KX);
#pragma unroll
    for (int kc = 0; kc < KCX; ++kc) {               // x part: B from LDS
      v16bf a = load_a_frag(XBc, KX, kc * 32, lane);
#pragma unroll
      for (int g4 = 0; g4 < 4; ++g4) {
        v16bf b = load_b_frag(Wl, KTOT, g4 * H_DIM + wv * 16,
                              H_DIM + kc * 32, lane);
        acc[g4] = __builtin_amdgcn_wmma_f32_16x16x32_bf16(
                      false, a, false, b, (short)0, acc[g4], false, false);
      }
    }
    __syncthreads();     // all reads of Ht done -> safe to overwrite h

    // ---- elementwise cell update (TRANS tanh, state in registers) ---------
#pragma unroll
    for (int r = 0; r < 8; ++r) {
      float gi = fast_sigmoid(acc[0][r]);
      float gf = fast_sigmoid(acc[1][r]);
      float gg = fast_tanh(acc[2][r]);
      float go = fast_sigmoid(acc[3][r]);
      float c2 = gf * c_st[r] + gi * gg;
      c_st[r]  = c2;
      float hn = go * fast_tanh(c2);
      h_f32[r] = hn;

      int m = r + mrow;
      __bf16 hb = to_bf16(hn);
      Ht[m * H_DIM + hcol] = hb;                                      // h(t+1)
      y_out[(size_t)t * BATCH * H_DIM + (size_t)(b0 + m) * H_DIM + hcol] = hb;
    }
    // next iteration's top barrier publishes these h writes
  }

  // Final h_n / c_n (f32): d_out tail is [L,B,H] h then [L,B,H] c.
#pragma unroll
  for (int r = 0; r < 8; ++r) {
    int m = r + mrow;
    size_t base = (size_t)layer * BATCH * H_DIM + (size_t)(b0 + m) * H_DIM + hcol;
    hc_out[base] = h_f32[r];
    hc_out[(size_t)2 * BATCH * H_DIM + base] = c_st[r];
  }
}

// ---------------------------------------------------------------------------
// Final FC: out[T*B,32] = y1[T*B,128] x fc_w[32,128]^T + fc_b. Bias is
// preloaded into the WMMA accumulator. 8 wave-tiles per block.
// ---------------------------------------------------------------------------
__global__ __launch_bounds__(256) void fc_kernel(
    const __bf16* __restrict__ y1,     // [T*B, H] bf16
    const float*  __restrict__ fc_w,   // [32, 128]
    const float*  __restrict__ fc_b,   // [32]
    float*        __restrict__ out)    // [T*B, 32]
{
  __shared__ __bf16 Wl[OUT_DIM * H_DIM];     // 8 KB
  const int tid  = threadIdx.x;
  const int lane = tid & 31;
  const int wv   = tid >> 5;

  for (int idx = tid; idx < OUT_DIM * H_DIM; idx += 256)
    Wl[idx] = to_bf16(fc_w[idx]);
  __syncthreads();

  const int gw    = blockIdx.x * 8 + wv;
  const int mtile = gw >> 1;
  const int ntile = gw & 1;
  const int n     = ntile * 16 + (lane & 15);

  float bias = fc_b[n];
  v8f acc;
#pragma unroll
  for (int r = 0; r < 8; ++r) acc[r] = bias;

  const __bf16* Abase = y1 + (size_t)mtile * 16 * H_DIM;
#pragma unroll
  for (int kc = 0; kc < H_DIM / 32; ++kc) {
    v16bf a = load_a_frag(Abase, H_DIM, kc * 32, lane);
    v16bf b = load_b_frag(Wl, H_DIM, ntile * 16, kc * 32, lane);
    acc = __builtin_amdgcn_wmma_f32_16x16x32_bf16(
              false, a, false, b, (short)0, acc, false, false);
  }

  const int mrow = (lane >> 4) << 3;
#pragma unroll
  for (int r = 0; r < 8; ++r) {
    int m = r + mrow;
    out[((size_t)mtile * 16 + m) * OUT_DIM + n] = acc[r];
  }
}

// ---------------------------------------------------------------------------
// Host launcher.
// d_out: out[T,B,32] then h_n[2,B,H] then c_n[2,B,H] (f32).
// d_ws:  x_bf16 (32MB) + y0 (64MB) + y1 (64MB), all bf16.
// ---------------------------------------------------------------------------
extern "C" void kernel_launch(void* const* d_in, const int* in_sizes, int n_in,
                              void* d_out, int out_size, void* d_ws, size_t ws_size,
                              hipStream_t stream) {
  (void)in_sizes; (void)n_in; (void)out_size; (void)ws_size;

  const float* x     = (const float*)d_in[0];
  const float* w_ih0 = (const float*)d_in[1];
  const float* w_hh0 = (const float*)d_in[2];
  const float* w_ih1 = (const float*)d_in[3];
  const float* w_hh1 = (const float*)d_in[4];
  const float* fc_w  = (const float*)d_in[5];
  const float* fc_b  = (const float*)d_in[6];

  float* out = (float*)d_out;
  float* hc  = out + (size_t)T_STEPS * BATCH * OUT_DIM;

  __bf16* x_bf = (__bf16*)d_ws;
  __bf16* y0   = x_bf + (size_t)T_STEPS * BATCH * IN_DIM;
  __bf16* y1   = y0   + (size_t)T_STEPS * BATCH * H_DIM;

  // Dynamic LDS: weights [512][KTOT] + h tile + x double buffer (bf16).
  const size_t shmem0 = (size_t)G_DIM * (H_DIM + IN_DIM) * 2 + 16 * H_DIM * 2
                      + 2 * (size_t)16 * IN_DIM * 2;          // ~205 KB
  const size_t shmem1 = (size_t)G_DIM * (H_DIM + H_DIM) * 2 + 16 * H_DIM * 2
                      + 2 * (size_t)16 * H_DIM * 2;           // ~274 KB
  (void)hipFuncSetAttribute(
      reinterpret_cast<const void*>(&lstm_scan_kernel<IN_DIM>),
      hipFuncAttributeMaxDynamicSharedMemorySize, (int)shmem0);
  (void)hipFuncSetAttribute(
      reinterpret_cast<const void*>(&lstm_scan_kernel<H_DIM>),
      hipFuncAttributeMaxDynamicSharedMemorySize, (int)shmem1);

  // 1) x -> bf16 (fully parallel).
  const int n4 = T_STEPS * BATCH * IN_DIM / 4;
  cvt_kernel<<<n4 / 256, 256, 0, stream>>>(x, x_bf, n4);

  // 2) Layer 0 scan, 3) layer 1 scan (stream-ordered).
  lstm_scan_kernel<IN_DIM><<<BATCH / 16, 256, shmem0, stream>>>(
      x_bf, w_ih0, w_hh0, y0, hc, 0);
  lstm_scan_kernel<H_DIM><<<BATCH / 16, 256, shmem1, stream>>>(
      y0, w_ih1, w_hh1, y1, hc, 1);

  // 4) FC over all T*B rows.
  const int fc_blocks = (T_STEPS * BATCH / 16) * 2 / 8;   // 4096
  fc_kernel<<<fc_blocks, 256, 0, stream>>>(y1, fc_w, fc_b, out);
}
